// Int8OPTAttention_87875030876541
// MI455X (gfx1250) — 8-bit matrix operand order repaired and checked by reading, compiled, not run on hardware
//
#include <hip/hip_runtime.h>
#include <cstdint>


typedef __attribute__((ext_vector_type(8))) int v8i;

#define TT   2048
#define DD   2048
#define HH   32
#define HDIM 64
#define MM   4096

static __device__ __forceinline__ v8i wmma_iu8(v8i a, v8i b, v8i c) {
  return __builtin_amdgcn_wmma_i32_16x16x64_iu8(true, a, true, b, c, false, false);
}

static __device__ __forceinline__ v8i load_a_frag(const int8_t* base, int ld) {
  const int lane = threadIdx.x & 31;
  const int m    = lane & 15;
  const int kh   = (lane >> 4) << 4;
  const int8_t* p = base + (size_t)m * ld + kh;
  int4 lo = *(const int4*)(p);
  int4 hi = *(const int4*)(p + 32);
  v8i a;
  a[0] = lo.x; a[1] = lo.y; a[2] = lo.z; a[3] = lo.w;
  a[4] = hi.x; a[5] = hi.y; a[6] = hi.z; a[7] = hi.w;
  return a;
}

static __device__ __forceinline__ v8i load_b_frag(const int8_t* base, int ld) {
  const int lane = threadIdx.x & 31;
  const int n    = lane & 15;
  const int kq   = (lane >> 4) << 4;
  const int8_t* p = base + (size_t)n * ld + kq;
  int4 lo = *(const int4*)(p);
  int4 hi = *(const int4*)(p + 32);
  v8i b;
  b[0] = lo.x; b[1] = lo.y; b[2] = lo.z; b[3] = lo.w;
  b[4] = hi.x; b[5] = hi.y; b[6] = hi.z; b[7] = hi.w;
  return b;
}

static __device__ __forceinline__ int sat8(float x) {
  float r = rintf(x);
  r = fminf(fmaxf(r, -128.f), 127.f);
  return (int)r;
}

template <int MODE>
__global__ __launch_bounds__(256) void gemm_i8_kernel(
    const int8_t* __restrict__ X, const int8_t* __restrict__ W,
    const void* __restrict__ bias, void* __restrict__ out, float alpha) {
  const int lane = threadIdx.x & 31;
  const int wave = threadIdx.x >> 5;
  const int wm = wave & 3;
  const int wn = wave >> 2;
  const int m0 = blockIdx.y * 128 + wm * 32;
  const int n0 = blockIdx.x * 64  + wn * 32;

  v8i acc[2][2];
  acc[0][0] = {}; acc[0][1] = {}; acc[1][0] = {}; acc[1][1] = {};

#pragma unroll 2
  for (int kc = 0; kc < DD; kc += 64) {
    v8i a0 = load_a_frag(X + (size_t)m0 * DD + kc, DD);
    v8i a1 = load_a_frag(X + (size_t)(m0 + 16) * DD + kc, DD);
    v8i b0 = load_b_frag(W + (size_t)n0 * DD + kc, DD);
    v8i b1 = load_b_frag(W + (size_t)(n0 + 16) * DD + kc, DD);
    acc[0][0] = wmma_iu8(a0, b0, acc[0][0]);
    acc[0][1] = wmma_iu8(a0, b1, acc[0][1]);
    acc[1][0] = wmma_iu8(a1, b0, acc[1][0]);
    acc[1][1] = wmma_iu8(a1, b1, acc[1][1]);
  }

  const int nloc = lane & 15;
  const int hgrp = lane >> 4;

  if constexpr (MODE == 0) {
    const int8_t* b8 = (const int8_t*)bias;
    int8_t* o8 = (int8_t*)out;
#pragma unroll
    for (int mi = 0; mi < 2; ++mi)
#pragma unroll
      for (int ni = 0; ni < 2; ++ni) {
        const int n = n0 + ni * 16 + nloc;
        const float bb = (float)b8[n];
#pragma unroll
        for (int r = 0; r < 8; ++r) {
          const int m = m0 + mi * 16 + 8 * hgrp + r;
          o8[(size_t)m * DD + n] = (int8_t)sat8(alpha * (float)acc[mi][ni][r] + bb);
        }
      }
  } else if constexpr (MODE == 1) {
    const int8_t* b8 = (const int8_t*)bias;
    int8_t* o8 = (int8_t*)out;
#pragma unroll
    for (int mi = 0; mi < 2; ++mi)
#pragma unroll
      for (int ni = 0; ni < 2; ++ni) {
        const int n = n0 + ni * 16 + nloc;
        const float bb = (float)b8[n];
        const int mstart = m0 + mi * 16 + 8 * hgrp;
        const int bidx = mstart >> 11;
        const int tstart = mstart & (TT - 1);
        uint32_t w0 = 0, w1 = 0;
#pragma unroll
        for (int r = 0; r < 4; ++r)
          w0 |= (uint32_t)(uint8_t)(int8_t)sat8(alpha * (float)acc[mi][ni][r] + bb) << (8 * r);
#pragma unroll
        for (int r = 4; r < 8; ++r)
          w1 |= (uint32_t)(uint8_t)(int8_t)sat8(alpha * (float)acc[mi][ni][r] + bb) << (8 * (r - 4));
        uint2 wv; wv.x = w0; wv.y = w1;
        int8_t* dst = o8 + ((size_t)((bidx * HH + (n >> 6)) * HDIM + (n & 63))) * TT + tstart;
        *(uint2*)dst = wv;
      }
  } else {
    const float* bf = (const float*)bias;
    float* of = (float*)out;
#pragma unroll
    for (int mi = 0; mi < 2; ++mi)
#pragma unroll
      for (int ni = 0; ni < 2; ++ni) {
        const int n = n0 + ni * 16 + nloc;
        const float bb = bf[n];
#pragma unroll
        for (int r = 0; r < 8; ++r) {
          const int m = m0 + mi * 16 + 8 * hgrp + r;
          of[(size_t)m * DD + n] = alpha * (float)acc[mi][ni][r] + bb;
        }
      }
  }
}

__global__ __launch_bounds__(256) void attn_kernel(
    const int8_t* __restrict__ q, const int8_t* __restrict__ k,
    const int8_t* __restrict__ vT, int8_t* __restrict__ o) {
  __shared__ __align__(16) int8_t plds[8][16 * 64];

  const int lane = threadIdx.x & 31;
  const int wave = threadIdx.x >> 5;
  const int bh = blockIdx.x >> 4;
  const int tb = blockIdx.x & 15;
  const int b  = bh >> 5;
  const int h  = bh & 31;
  const int t0 = tb * 128 + wave * 16;

  const int nloc = lane & 15;
  const int hgrp = lane >> 4;

  const int8_t* qbase = q + (size_t)(b * TT + t0) * DD + h * HDIM;
  const v8i qa = load_a_frag(qbase, DD);

  const int8_t* kbh = k  + (size_t)b * TT * DD + h * HDIM;
  const int8_t* vbh = vT + (size_t)bh * HDIM * TT;

  v8i pv[4];
  pv[0] = {}; pv[1] = {}; pv[2] = {}; pv[3] = {};

  float lsum[8];
#pragma unroll
  for (int r = 0; r < 8; ++r) lsum[r] = 0.f;

  const int nchunk = (t0 + 16 + 63) >> 6;

  for (int c = 0; c < nchunk; ++c) {
    const int s0 = c * 64;
    v8i cc[4];
#pragma unroll
    for (int sub = 0; sub < 4; ++sub) {
      v8i bf = load_b_frag(kbh + (size_t)(s0 + sub * 16) * DD, DD);
      v8i z = {};
      cc[sub] = wmma_iu8(qa, bf, z);
    }
#pragma unroll
    for (int sub = 0; sub < 4; ++sub) {
      const int s = s0 + sub * 16 + nloc;
#pragma unroll
      for (int r = 0; r < 8; ++r) {
        const int t = t0 + r + 8 * hgrp;
        float e = __expf(2e-5f * (float)cc[sub][r]);
        e = (s <= t) ? e : 0.f;
        lsum[r] += e;
      }
    }
  }
#pragma unroll
  for (int off = 1; off < 16; off <<= 1)
#pragma unroll
    for (int r = 0; r < 8; ++r)
      lsum[r] += __shfl_xor(lsum[r], off, 32);

  float rq[8];
#pragma unroll
  for (int r = 0; r < 8; ++r) rq[r] = 127.0f / lsum[r];

  int8_t* myp = plds[wave];
  for (int c = 0; c < nchunk; ++c) {
    const int s0 = c * 64;
    v8i cc[4];
#pragma unroll
    for (int sub = 0; sub < 4; ++sub) {
      v8i bf = load_b_frag(kbh + (size_t)(s0 + sub * 16) * DD, DD);
      v8i z = {};
      cc[sub] = wmma_iu8(qa, bf, z);
    }
#pragma unroll
    for (int sub = 0; sub < 4; ++sub) {
      const int s = s0 + sub * 16 + nloc;
#pragma unroll
      for (int r = 0; r < 8; ++r) {
        const int t = t0 + r + 8 * hgrp;
        float e = __expf(2e-5f * (float)cc[sub][r]);
        e = (s <= t) ? e : 0.f;
        myp[(r + 8 * hgrp) * 64 + sub * 16 + nloc] = (int8_t)sat8(e * rq[r]);
      }
    }
    v8i pa = load_a_frag(myp, 64);
#pragma unroll
    for (int sub = 0; sub < 4; ++sub) {
      v8i vb = load_b_frag(vbh + (size_t)(sub * 16) * TT + s0, TT);
      pv[sub] = wmma_iu8(pa, vb, pv[sub]);
    }
  }

  int8_t* obase = o + (size_t)(b * TT + t0) * DD + h * HDIM;
#pragma unroll
  for (int sub = 0; sub < 4; ++sub)
#pragma unroll
    for (int r = 0; r < 8; ++r) {
      const int m = r + 8 * hgrp;
      obase[(size_t)m * DD + sub * 16 + nloc] =
          (int8_t)sat8(1e-2f * (float)pv[sub][r]);
    }
}

extern "C" void kernel_launch(void* const* d_in, const int* in_sizes, int n_in,
                              void* d_out, int out_size, void* d_ws, size_t ws_size,
                              hipStream_t stream) {
  (void)in_sizes; (void)n_in; (void)out_size; (void)ws_size;
  const int8_t* hs = (const int8_t*)d_in[0];
  const int8_t* Wq = (const int8_t*)d_in[2];
  const int8_t* bq = (const int8_t*)d_in[3];
  const int8_t* Wk = (const int8_t*)d_in[4];
  const int8_t* bk = (const int8_t*)d_in[5];
  const int8_t* Wv = (const int8_t*)d_in[6];
  const int8_t* bv = (const int8_t*)d_in[7];
  const int8_t* Wo = (const int8_t*)d_in[8];
  const float*  bo = (const float*)d_in[9];
  float* out = (float*)d_out;

  const size_t SZ = (size_t)MM * DD;
  int8_t* ws = (int8_t*)d_ws;
  int8_t* qb = ws;
  int8_t* kb = ws + SZ;
  int8_t* vT = ws + 2 * SZ;
  int8_t* ao = ws + 3 * SZ;

  dim3 gg(DD / 64, MM / 128);
  dim3 gb(256);

  gemm_i8_kernel<0><<<gg, gb, 0, stream>>>(hs, Wq, bq, qb, 3e-4f);
  gemm_i8_kernel<0><<<gg, gb, 0, stream>>>(hs, Wk, bk, kb, 3e-4f);
  gemm_i8_kernel<1><<<gg, gb, 0, stream>>>(hs, Wv, bv, vT, 3e-4f);
  attn_kernel<<<dim3(2 * HH * (TT / 128)), gb, 0, stream>>>(qb, kb, vT, ao);
  gemm_i8_kernel<2><<<gg, gb, 0, stream>>>(ao, Wo, bo, out, 1e-4f);
}
